// TransformerBlock_60799557042378
// MI455X (gfx1250) — matrix operand order repaired and checked by reading, not compiled, not run on hardware
//
#include <hip/hip_runtime.h>
#include <hip/hip_bf16.h>
#include <cstdint>

typedef _Float16 f16;
typedef _Float16 v16h __attribute__((ext_vector_type(16)));
typedef float    v8f  __attribute__((ext_vector_type(8)));
typedef unsigned int u32x4 __attribute__((ext_vector_type(4)));
typedef int          i32x8 __attribute__((ext_vector_type(8)));
typedef int          i32x4 __attribute__((ext_vector_type(4)));

#define B_   2
#define S_   2048
#define DIM_ 1024
#define H_   16
#define DH_  64
#define MLP_ 4096
#define ROWS (B_*S_)   // 4096

// 16-bit A-matrix 16x32 layout (contiguous, matches B fills): lane group g = lane/16.
// VGPR v:  K = 2v + 16g  (each dword = K,K+1 pair)
static __device__ __forceinline__ int a_koff(int v, int g) {
    return 2*v + 16*g;
}

// ---------------------------------------------------------------------------
// Tensor Data Mover: 2D tile load Global -> LDS (D# per ISA ch.8).
// data_size = 4 bytes; dims/strides in dword units; rows packed tight in LDS.
// ---------------------------------------------------------------------------
static __device__ __forceinline__ void tdm_load_2d(uint32_t lds_off,
                                                   const void* gaddr,
                                                   uint32_t tensor_d0,   // row length (dwords)
                                                   uint32_t tensor_d1,   // #rows in tensor
                                                   uint32_t tile_d0,     // tile row chunk (dwords)
                                                   uint32_t tile_d1,     // tile #rows
                                                   uint64_t stride_d0) { // row stride (dwords)
    uint64_t ga = (uint64_t)(uintptr_t)gaddr;
    u32x4 g0;
    g0[0] = 1u;                                   // count=1, user desc, gather off
    g0[1] = lds_off;                              // lds_addr (bytes)
    g0[2] = (uint32_t)ga;                         // global_addr[31:0]
    g0[3] = (uint32_t)(ga >> 32) | (2u << 30);    // global_addr[56:32] | type=2
    i32x8 g1;
    g1[0] = (int)(2u << 16);                      // wg_mask=0, data_size=2 (4B)
    g1[1] = (int)((tensor_d0 & 0xffffu) << 16);   // tdim0[15:0] in bits[63:48]
    g1[2] = (int)((tensor_d0 >> 16) | ((tensor_d1 & 0xffffu) << 16));
    g1[3] = (int)((tensor_d1 >> 16) | (tile_d0 << 16));  // tile_dim0 bits[127:112]
    g1[4] = (int)(tile_d1 & 0xffffu);             // tile_dim1; tile_dim2=0
    g1[5] = (int)(uint32_t)stride_d0;             // tensor_dim0_stride[31:0]
    g1[6] = (int)(uint32_t)(stride_d0 >> 32);     // stride[47:32]; dim1_stride[15:0]=0
    g1[7] = 0;
    i32x4 z4 = {};
#if __clang_major__ >= 23
    i32x8 z8 = {};
    __builtin_amdgcn_tensor_load_to_lds(g0, g1, z4, z4, z8, 0);
#else
    __builtin_amdgcn_tensor_load_to_lds(g0, g1, z4, z4, 0);
#endif
}

// ---------------------------------------------------------------------------
// Weight cast + transpose: fp32 W[K,N] -> f16 Wt[N,K]
// ---------------------------------------------------------------------------
__global__ __launch_bounds__(256) void cast_wT(const float* __restrict__ W,
                                               f16* __restrict__ Wt, int K, int N) {
    size_t i = (size_t)blockIdx.x * 256 + threadIdx.x;
    if (i < (size_t)K * N) {
        int k = (int)(i / N), n = (int)(i % N);
        Wt[(size_t)n * K + k] = (f16)W[i];
    }
}

// ---------------------------------------------------------------------------
// LayerNorm fp32 [R, DIM] -> f16 [R, DIM]
// ---------------------------------------------------------------------------
__global__ __launch_bounds__(256) void ln_kernel(const float* __restrict__ x,
                                                 const float* __restrict__ gm,
                                                 const float* __restrict__ bt,
                                                 f16* __restrict__ out) {
    __shared__ float red[256];
    int r = blockIdx.x;
    const float* xr = x + (size_t)r * DIM_;
    float s = 0.f, sq = 0.f;
    for (int i = threadIdx.x; i < DIM_; i += 256) { float v = xr[i]; s += v; sq += v * v; }
    red[threadIdx.x] = s; __syncthreads();
    for (int o = 128; o > 0; o >>= 1) { if ((int)threadIdx.x < o) red[threadIdx.x] += red[threadIdx.x + o]; __syncthreads(); }
    float mean = red[0] / DIM_;
    __syncthreads();
    red[threadIdx.x] = sq; __syncthreads();
    for (int o = 128; o > 0; o >>= 1) { if ((int)threadIdx.x < o) red[threadIdx.x] += red[threadIdx.x + o]; __syncthreads(); }
    float var = red[0] / DIM_ - mean * mean;
    float inv = rsqrtf(var + 1e-5f);
    for (int i = threadIdx.x; i < DIM_; i += 256) {
        float v = (xr[i] - mean) * inv * gm[i] + bt[i];
        out[(size_t)r * DIM_ + i] = (f16)v;
    }
}

// ---------------------------------------------------------------------------
// WMMA GEMM with double-buffered TDM-staged LDS tiles.
// C[M,N] = A[M,K](f16,row-major) @ Bt[N,K](f16, B pre-transposed).
// Block: 256 threads = 8 waves; wave tile 32x32 (2x2 WMMA); block tile 128x64.
// Pipeline: wave0 preloads stage0 via tensor_load_to_lds; each K-step it
// issues the NEXT stage's 2 DMAs, then s_wait_tensorcnt<=2 (in-order TDM ->
// current stage landed) so DMA overlaps the WMMAs of the current stage.
// EPI: 0 = fp32 out; 1 = +res; 2 = +bias+res; 3 = +bias, exact GELU, f16 out
// ---------------------------------------------------------------------------
template<int EPI>
__global__ __launch_bounds__(256) void gemm_wmma(const f16* __restrict__ A,
                                                 const f16* __restrict__ Bt,
                                                 const float* __restrict__ bias,
                                                 const float* __restrict__ res,
                                                 float* __restrict__ outF,
                                                 f16* __restrict__ outH,
                                                 int M, int N, int K) {
    __shared__ f16 Atile[2][128 * 32];   // 2 x 8 KB
    __shared__ f16 Btile[2][64 * 32];    // 2 x 4 KB
    const int lane = threadIdx.x & 31;
    const int wid  = threadIdx.x >> 5;
    const int wr = wid & 3, wc = wid >> 2;
    const int m0 = blockIdx.x * 128;
    const int n0 = blockIdx.y * 64;
    const int m0l = wr * 32, n0l = wc * 32;
    const int g  = lane >> 4;
    const int lm = lane & 15;

    const uint32_t aoff0 = (uint32_t)(size_t)(uintptr_t)&Atile[0][0];
    const uint32_t aoff1 = (uint32_t)(size_t)(uintptr_t)&Atile[1][0];
    const uint32_t boff0 = (uint32_t)(size_t)(uintptr_t)&Btile[0][0];
    const uint32_t boff1 = (uint32_t)(size_t)(uintptr_t)&Btile[1][0];

    union U { v16h v; uint32_t u[8]; };
    v8f c[2][2] = {};

    // preload stage 0 (kk = 0)
    if (wid == 0) {
        tdm_load_2d(aoff0, A + (size_t)m0 * K, (uint32_t)(K / 2), (uint32_t)M,
                    16u, 128u, (uint64_t)(K / 2));
        tdm_load_2d(boff0, Bt + (size_t)n0 * K, (uint32_t)(K / 2), (uint32_t)N,
                    16u, 64u, (uint64_t)(K / 2));
    }

    for (int kk = 0; kk < K; kk += 32) {
        const int s = (kk >> 5) & 1;
        if (wid == 0) {
            if (kk + 32 < K) {
                // issue next stage's DMAs, then wait only for current stage
                tdm_load_2d(s ? aoff0 : aoff1, A + (size_t)m0 * K + kk + 32,
                            (uint32_t)(K / 2), (uint32_t)M, 16u, 128u, (uint64_t)(K / 2));
                tdm_load_2d(s ? boff0 : boff1, Bt + (size_t)n0 * K + kk + 32,
                            (uint32_t)(K / 2), (uint32_t)N, 16u, 64u, (uint64_t)(K / 2));
                __builtin_amdgcn_s_wait_tensorcnt(2);
            } else {
                __builtin_amdgcn_s_wait_tensorcnt(0);
            }
        }
        __syncthreads();                 // stage s visible to all waves

        const uint32_t* A32 = (const uint32_t*)&Atile[s][0];
        const uint32_t* B32 = (const uint32_t*)&Btile[s][0];
        U a[2], bf[2];
#pragma unroll
        for (int t = 0; t < 2; ++t) {
            int row = m0l + t * 16 + lm;
#pragma unroll
            for (int v = 0; v < 8; ++v) a[t].u[v] = A32[(row * 32 + a_koff(v, g)) >> 1];
            int col = n0l + t * 16 + lm;
#pragma unroll
            for (int v = 0; v < 8; ++v) bf[t].u[v] = B32[(col * 32 + 16 * g + 2 * v) >> 1];
        }
#pragma unroll
        for (int i = 0; i < 2; ++i)
#pragma unroll
            for (int j = 0; j < 2; ++j)
                c[i][j] = __builtin_amdgcn_wmma_f32_16x16x32_f16(
                    false, a[i].v, false, bf[j].v, (short)0, c[i][j], false, false);
        __syncthreads();                 // all reads of stage s done before overwrite
    }

#pragma unroll
    for (int i = 0; i < 2; ++i)
#pragma unroll
        for (int j = 0; j < 2; ++j)
#pragma unroll
            for (int v = 0; v < 8; ++v) {
                int row = m0 + m0l + i * 16 + v + 8 * g;  // C/D: M = v + 8*(lane/16)
                int col = n0 + n0l + j * 16 + lm;         //      N = lane%16
                size_t idx = (size_t)row * N + col;
                float val = c[i][j][v];
                if (EPI == 1)      { val += res[idx]; outF[idx] = val; }
                else if (EPI == 2) { val += bias[col] + res[idx]; outF[idx] = val; }
                else if (EPI == 3) { val += bias[col];
                                     val = 0.5f * val * (1.f + erff(val * 0.70710678f));
                                     outH[idx] = (f16)val; }
                else               { outF[idx] = val; }
            }
}

// ---------------------------------------------------------------------------
// RoPE + split: qkv fp32 [B,S,H,3*DH] -> q(scaled by dh^-0.5),k f16 [B,H,S,DH],
// v transposed f16 [B,H,DH,S]. One 32-thread block per (b,h,s); thread = pair.
// ---------------------------------------------------------------------------
__global__ __launch_bounds__(32) void rope_kernel(const float* __restrict__ qkv,
                                                  f16* __restrict__ q,
                                                  f16* __restrict__ k,
                                                  f16* __restrict__ vT) {
    int s  = blockIdx.x % S_;
    int bh = blockIdx.x / S_;
    int h  = bh % H_;
    int b  = bh / H_;
    int i  = threadIdx.x;                       // pair index 0..31
    const float* row = qkv + ((size_t)(b * S_ + s)) * (3 * DIM_) + (size_t)h * (3 * DH_);
    float inv_freq = __powf(10000.f, -(float)(2 * i) / (float)DH_);
    float th = (float)s * inv_freq;
    float sn, cs; __sincosf(th, &sn, &cs);
    size_t base = ((size_t)bh * S_ + s) * DH_;

    float x0 = row[2 * i], x1 = row[2 * i + 1];
    q[base + 2 * i]     = (f16)(0.125f * (x0 * cs - x1 * sn));
    q[base + 2 * i + 1] = (f16)(0.125f * (x1 * cs + x0 * sn));
    x0 = row[DH_ + 2 * i]; x1 = row[DH_ + 2 * i + 1];
    k[base + 2 * i]     = (f16)(x0 * cs - x1 * sn);
    k[base + 2 * i + 1] = (f16)(x1 * cs + x0 * sn);
    x0 = row[2 * DH_ + 2 * i]; x1 = row[2 * DH_ + 2 * i + 1];
    size_t tb = (size_t)bh * DH_ * S_;
    vT[tb + (size_t)(2 * i)     * S_ + s] = (f16)x0;
    vT[tb + (size_t)(2 * i + 1) * S_ + s] = (f16)x1;
}

// ---------------------------------------------------------------------------
// Flash attention: wave = 16 q rows; block = 4 waves = 64 q rows of one (b,h).
// K-tiles of 32 keys: 4 WMMAs QK^T (dh=64 contraction) + 4 WMMAs P@V.
// Online softmax with width-16 shfl reductions (row lives in 16 lanes).
// ---------------------------------------------------------------------------
__global__ __launch_bounds__(128) void flash_kernel(const f16* __restrict__ q,
                                                    const f16* __restrict__ k,
                                                    const f16* __restrict__ vT,
                                                    const float* __restrict__ bias,
                                                    const int* __restrict__ pad,
                                                    f16* __restrict__ attn) {
    __shared__ f16 plds[4][16 * 32];
    const int lane = threadIdx.x & 31;
    const int wid  = threadIdx.x >> 5;
    const int g = lane >> 4, lm = lane & 15;

    const int nqb = S_ / 64;
    int qb = blockIdx.x % nqb;
    int bh = blockIdx.x / nqb;
    int h  = bh % H_;
    int b  = bh / H_;
    int q0 = qb * 64 + wid * 16;

    const uint32_t* q32 = (const uint32_t*)(q  + (size_t)bh * S_  * DH_);
    const uint32_t* k32 = (const uint32_t*)(k  + (size_t)bh * S_  * DH_);
    const uint32_t* v32 = (const uint32_t*)(vT + (size_t)bh * DH_ * S_);
    const float* biasb = bias + (size_t)b * S_ * S_;
    const int*   padb  = pad + b * S_;

    union U { v16h v; uint32_t u[8]; };
    U qf[2];
#pragma unroll
    for (int cch = 0; cch < 2; ++cch) {
        const uint32_t* ap = q32 + ((size_t)(q0 + lm) * DH_ + 32 * cch) / 2;
#pragma unroll
        for (int v = 0; v < 8; ++v) qf[cch].u[v] = ap[a_koff(v, g) >> 1];
    }

    float m[8], l[8];
#pragma unroll
    for (int v = 0; v < 8; ++v) { m[v] = -3e38f; l[v] = 0.f; }
    v8f o[4] = {};

    for (int kk0 = 0; kk0 < S_; kk0 += 32) {
        // prefetch next K/V tiles into cache (global_prefetch_b8)
        if (kk0 + 32 < S_) {
            __builtin_prefetch((const void*)(k32 + ((size_t)(kk0 + 32 + lm) * DH_) / 2), 0, 0);
            __builtin_prefetch((const void*)(v32 + ((size_t)lm * S_ + kk0 + 32) / 2), 0, 0);
        }
        v8f sc[2] = {};
#pragma unroll
        for (int sub = 0; sub < 2; ++sub)
#pragma unroll
            for (int cch = 0; cch < 2; ++cch) {
                U kf;
                const uint32_t* kp = k32 +
                    ((size_t)(kk0 + sub * 16 + lm) * DH_ + 32 * cch + 16 * g) / 2;
#pragma unroll
                for (int v = 0; v < 8; ++v) kf.u[v] = kp[v];
                sc[sub] = __builtin_amdgcn_wmma_f32_16x16x32_f16(
                    false, qf[cch].v, false, kf.v, (short)0, sc[sub], false, false);
            }
        // bias + key-padding mask (q pre-scaled by dh^-0.5)
#pragma unroll
        for (int sub = 0; sub < 2; ++sub) {
            int col = kk0 + sub * 16 + lm;
            float madd = padb[col] ? -3e38f : 0.f;
#pragma unroll
            for (int v = 0; v < 8; ++v) {
                int row = q0 + v + 8 * g;
                sc[sub][v] += biasb[(size_t)row * S_ + col] + madd;
            }
        }
        // online softmax update
#pragma unroll
        for (int v = 0; v < 8; ++v) {
            float r = fmaxf(sc[0][v], sc[1][v]);
#pragma unroll
            for (int off = 1; off < 16; off <<= 1) r = fmaxf(r, __shfl_xor(r, off, 16));
            float nm = fmaxf(m[v], r);
            float alpha = __expf(m[v] - nm);
            m[v] = nm;
            float p0 = __expf(sc[0][v] - nm);
            float p1 = __expf(sc[1][v] - nm);
            sc[0][v] = p0; sc[1][v] = p1;
            float rs = p0 + p1;
#pragma unroll
            for (int off = 1; off < 16; off <<= 1) rs += __shfl_xor(rs, off, 16);
            l[v] = l[v] * alpha + rs;
#pragma unroll
            for (int t = 0; t < 4; ++t) o[t][v] *= alpha;
        }
        // C-layout P -> LDS row-major f16 [16][32] -> reload as A-fragment
#pragma unroll
        for (int sub = 0; sub < 2; ++sub)
#pragma unroll
            for (int v = 0; v < 8; ++v)
                plds[wid][(v + 8 * g) * 32 + sub * 16 + lm] = (f16)sc[sub][v];
        __syncthreads();
        U pf;
        const uint32_t* pl = (const uint32_t*)&plds[wid][0];
#pragma unroll
        for (int v = 0; v < 8; ++v) pf.u[v] = pl[(lm * 32 + a_koff(v, g)) >> 1];
        __syncthreads();
        // O += P @ V  (V as B 32x16 from vT[B,H,DH,S]: pairs contiguous in keys)
#pragma unroll
        for (int t = 0; t < 4; ++t) {
            U vf;
            const uint32_t* vp = v32 + ((size_t)(t * 16 + lm) * S_ + kk0 + 16 * g) / 2;
#pragma unroll
            for (int v = 0; v < 8; ++v) vf.u[v] = vp[v];
            o[t] = __builtin_amdgcn_wmma_f32_16x16x32_f16(
                false, pf.v, false, vf.v, (short)0, o[t], false, false);
        }
    }
    // normalize + store f16 [B,S,H*DH]
#pragma unroll
    for (int t = 0; t < 4; ++t)
#pragma unroll
        for (int v = 0; v < 8; ++v) {
            int row = q0 + v + 8 * g;
            float val = o[t][v] / l[v];
            attn[((size_t)b * S_ + row) * DIM_ + h * DH_ + t * 16 + lm] = (f16)val;
        }
}

// ---------------------------------------------------------------------------
// Host-side orchestration
// ---------------------------------------------------------------------------
extern "C" void kernel_launch(void* const* d_in, const int* in_sizes, int n_in,
                              void* d_out, int out_size, void* d_ws, size_t ws_size,
                              hipStream_t stream) {
    const float* x     = (const float*)d_in[0];
    const float* bias  = (const float*)d_in[1];
    const float* ln1_g = (const float*)d_in[2];
    const float* ln1_b = (const float*)d_in[3];
    const float* W_qkv = (const float*)d_in[4];
    const float* W_o   = (const float*)d_in[5];
    const float* ln2_g = (const float*)d_in[6];
    const float* ln2_b = (const float*)d_in[7];
    const float* W1    = (const float*)d_in[8];
    const float* b1    = (const float*)d_in[9];
    const float* W2    = (const float*)d_in[10];
    const float* b2    = (const float*)d_in[11];
    const int*   pad   = (const int*)d_in[12];

    char* ws = (char*)d_ws;
    f16*   x1      = (f16*)ws;   ws += (size_t)ROWS * DIM_ * 2;
    f16*   WqkvT   = (f16*)ws;   ws += (size_t)(3 * DIM_) * DIM_ * 2;
    f16*   WoT     = (f16*)ws;   ws += (size_t)DIM_ * DIM_ * 2;
    f16*   W1T     = (f16*)ws;   ws += (size_t)MLP_ * DIM_ * 2;
    f16*   W2T     = (f16*)ws;   ws += (size_t)DIM_ * MLP_ * 2;
    float* qkv     = (float*)ws; ws += (size_t)ROWS * (3 * DIM_) * 4;
    f16*   qh      = (f16*)ws;   ws += (size_t)B_ * H_ * S_ * DH_ * 2;
    f16*   kh      = (f16*)ws;   ws += (size_t)B_ * H_ * S_ * DH_ * 2;
    f16*   vT      = (f16*)ws;   ws += (size_t)B_ * H_ * DH_ * S_ * 2;
    f16*   attn    = (f16*)ws;   ws += (size_t)ROWS * DIM_ * 2;
    float* x_after = (float*)ws; ws += (size_t)ROWS * DIM_ * 4;
    f16*   x2      = (f16*)ws;   ws += (size_t)ROWS * DIM_ * 2;
    f16*   hbuf    = (f16*)ws;   ws += (size_t)ROWS * MLP_ * 2;

    // 1) weights -> f16 transposed
    cast_wT<<<(DIM_ * 3 * DIM_ + 255) / 256, 256, 0, stream>>>(W_qkv, WqkvT, DIM_, 3 * DIM_);
    cast_wT<<<(DIM_ * DIM_     + 255) / 256, 256, 0, stream>>>(W_o,   WoT,   DIM_, DIM_);
    cast_wT<<<(DIM_ * MLP_     + 255) / 256, 256, 0, stream>>>(W1,    W1T,   DIM_, MLP_);
    cast_wT<<<(MLP_ * DIM_     + 255) / 256, 256, 0, stream>>>(W2,    W2T,   MLP_, DIM_);

    // 2) LN1 -> f16
    ln_kernel<<<ROWS, 256, 0, stream>>>(x, ln1_g, ln1_b, x1);

    // 3) QKV projection (fp32 out)
    gemm_wmma<0><<<dim3(ROWS / 128, (3 * DIM_) / 64), 256, 0, stream>>>(
        x1, WqkvT, nullptr, nullptr, qkv, nullptr, ROWS, 3 * DIM_, DIM_);

    // 4) RoPE + split + V transpose
    rope_kernel<<<B_ * H_ * S_, 32, 0, stream>>>(qkv, qh, kh, vT);

    // 5) flash attention
    flash_kernel<<<B_ * H_ * (S_ / 64), 128, 0, stream>>>(qh, kh, vT, bias, pad, attn);

    // 6) O projection + residual (fp32 out)
    gemm_wmma<1><<<dim3(ROWS / 128, DIM_ / 64), 256, 0, stream>>>(
        attn, WoT, nullptr, x, x_after, nullptr, ROWS, DIM_, DIM_);

    // 7) LN2 -> f16
    ln_kernel<<<ROWS, 256, 0, stream>>>(x_after, ln2_g, ln2_b, x2);

    // 8) MLP up + GELU (f16 out)
    gemm_wmma<3><<<dim3(ROWS / 128, MLP_ / 64), 256, 0, stream>>>(
        x2, W1T, b1, nullptr, nullptr, hbuf, ROWS, MLP_, DIM_);

    // 9) MLP down + bias + residual -> d_out (fp32)
    gemm_wmma<2><<<dim3(ROWS / 128, DIM_ / 64), 256, 0, stream>>>(
        hbuf, W2T, b2, x_after, (float*)d_out, nullptr, ROWS, DIM_, MLP_);
}
